// dynedge_15899968930238
// MI455X (gfx1250) — matrix operand order repaired and checked by reading, not compiled, not run on hardware
//
#include <hip/hip_runtime.h>

typedef __bf16 bf16;
typedef __attribute__((ext_vector_type(16))) __bf16 v16bf;
typedef __attribute__((ext_vector_type(8)))  __bf16 v8bf;
typedef __attribute__((ext_vector_type(8)))  float  v8f;
typedef __attribute__((ext_vector_type(4)))  unsigned int v4u;
typedef __attribute__((ext_vector_type(8)))  int v8i;
typedef __attribute__((ext_vector_type(4)))  int v4i;

static __device__ __forceinline__ float leaky_f(float x) {
  return x > 0.f ? x : 0.01f * x;
}

// A-fragment (16x32 bf16, M x K) in this code's contiguous k order (B uses the same):
// lanes 0-15 hold K = 0..15, lanes 16-31 hold K = 16..31.
// p must already include +half*16 elements.
static __device__ __forceinline__ v16bf load_a_frag(const bf16* p) {
  union U { v16bf v; v8bf h[2]; } u;
  u.h[0] = *(const v8bf*)(p);
  u.h[1] = *(const v8bf*)(p + 8);
  return u.v;
}

// B-fragment (32x16 bf16, K x N): lane n holds column n; lanes 0-15 K=0..15,
// lanes 16-31 K=16..31 -> 32 contiguous bf16 of a transposed-weight row.
// p must already include +half*16 elements.
static __device__ __forceinline__ v16bf load_b_frag(const bf16* p) {
  return *(const v16bf*)(p);
}

// ---------------------------------------------------------------------------
// TDM: async-load a 2D tile (32 contiguous bf16 x `rows` rows, row pitch
// `stride_elems`) from global into LDS at byte offset `lds_off` (row-major,
// 64B per row).  D# per CDNA5 ISA sec. 8.3/8.4; groups 2/3 unused (2D tile).
// Tracked by TENSORcnt; caller pairs with s_wait_tensorcnt + barrier.
// This toolchain exposes the 6-arg builtin (extra v8i group before cpol).
// ---------------------------------------------------------------------------
static __device__ __forceinline__ void tdm_load_2d(const bf16* gsrc,
                                                   unsigned int lds_off,
                                                   unsigned int rows,
                                                   unsigned int stride_elems) {
  const unsigned long long ga = (unsigned long long)(uintptr_t)gsrc;
  v4u g0;
  g0[0] = 1u;                                   // count=1, user mode
  g0[1] = lds_off;                              // lds_addr (bytes)
  g0[2] = (unsigned int)ga;                     // global_addr[31:0]
  g0[3] = (unsigned int)((ga >> 32) & 0x1FFFFFFu) | (2u << 30);  // [56:32]|type=2
  v8i g1;
  g1[0] = (int)(1u << 16);          // workgroup_mask=0, data_size=1 (2 bytes)
  g1[1] = (int)(32u << 16);         // tensor_dim0 = 32 (bits 79:48, low half)
  g1[2] = (int)(rows << 16);        // tensor_dim0 hi=0 | tensor_dim1 lo16
  g1[3] = (int)(32u << 16);         // tensor_dim1 hi=0 | tile_dim0 = 32
  g1[4] = (int)rows;                // tile_dim1 = rows, tile_dim2 = 0
  g1[5] = (int)stride_elems;        // tensor_dim0_stride (data_size units)
  g1[6] = 0;                        // stride hi | tensor_dim1_stride lo
  g1[7] = 0;
  const v4i z4 = {0, 0, 0, 0};
  const v8i z8 = {0, 0, 0, 0, 0, 0, 0, 0};
  __builtin_amdgcn_tensor_load_to_lds(g0, g1, z4, z4, z8, 0);
}

// ---------------------------------------------------------------------------
// Weight prep: W (Kin x Nin, f32 row-major) -> WT (Npad x Kpad, bf16), zero pad
// ---------------------------------------------------------------------------
__global__ void pack_wT(const float* __restrict__ W, bf16* __restrict__ WT,
                        int Kin, int Nin, int Kpad, int Npad) {
  int t = blockIdx.x * 256 + threadIdx.x;
  if (t >= Kpad * Npad) return;
  int n = t / Kpad;
  int k = t - n * Kpad;
  float v = (n < Nin && k < Kin) ? W[(size_t)k * Nin + n] : 0.f;
  WT[t] = (bf16)v;
}

__global__ void pack_bias(const float* __restrict__ B, float* __restrict__ BP,
                          int Nin, int Npad) {
  int t = blockIdx.x * 256 + threadIdx.x;
  if (t < Npad) BP[t] = (t < Nin) ? B[t] : 0.f;
}

// ---------------------------------------------------------------------------
// KNN: one block per batch, thread i = point i, top-4 smallest distances
// over first 3 feature channels (self excluded, ties -> lower index).
// ---------------------------------------------------------------------------
__global__ __launch_bounds__(256) void knn_kernel(const float* __restrict__ X,
                                                  int stride,
                                                  int* __restrict__ IDX) {
  __shared__ float px[256], py[256], pz[256];
  const int bb = blockIdx.x;
  const int i  = threadIdx.x;
  const float* row = X + ((size_t)bb * 256 + i) * stride;
  px[i] = row[0]; py[i] = row[1]; pz[i] = row[2];
  __syncthreads();
  const float xi = px[i], yi = py[i], zi = pz[i];
  float bd0 = 1e30f, bd1 = 1e30f, bd2 = 1e30f, bd3 = 1e30f;
  int   bi0 = 0, bi1 = 0, bi2 = 0, bi3 = 0;
  for (int j = 0; j < 256; ++j) {
    if (j == i) continue;
    const float dx = px[j] - xi, dy = py[j] - yi, dz = pz[j] - zi;
    const float d  = dx * dx + dy * dy + dz * dz;
    if (d < bd3) {
      if (d < bd0)      { bd3=bd2; bi3=bi2; bd2=bd1; bi2=bi1; bd1=bd0; bi1=bi0; bd0=d; bi0=j; }
      else if (d < bd1) { bd3=bd2; bi3=bi2; bd2=bd1; bi2=bi1; bd1=d;  bi1=j; }
      else if (d < bd2) { bd3=bd2; bi3=bi2; bd2=d;  bi2=j; }
      else              { bd3=d;  bi3=j; }
    }
  }
  int* o = IDX + ((size_t)bb * 256 + i) * 4;
  o[0] = bi0; o[1] = bi1; o[2] = bi2; o[3] = bi3;
}

// ---------------------------------------------------------------------------
// EdgeConv: block = (batch, group of 16 points) -> 64 edge rows.
//   E(64 x KE) @ W1T^T (+b1, leaky) -> H1(64 x HID) in LDS (bf16)
//   H1 @ W2T^T (+b2, leaky) -> sum over 4 neighbors -> Y(16 x OUT) f32
// 8 waves: mt = wave&3 (16-row strip), wn = wave>>2 (N half).
// Weight K-slices (32 x N) are TDM-staged into LDS, double buffered:
// wave 0 issues tensor_load_to_lds, waits TENSORcnt, barrier publishes.
// ---------------------------------------------------------------------------
template <int CIN, int KE, int HID, int OUT>
__global__ __launch_bounds__(256) void edgeconv_kernel(
    const float* __restrict__ X, const int* __restrict__ IDX,
    const bf16* __restrict__ W1T, const float* __restrict__ B1,
    const bf16* __restrict__ W2T, const float* __restrict__ B2,
    float* __restrict__ Y) {
  constexpr int MT   = 64;
  constexpr int NT1W = (HID / 16) / 2;
  constexpr int NT2W = (OUT / 16) / 2;
  constexpr int NK1  = KE / 32;
  constexpr int NK2  = HID / 32;

  extern __shared__ char smem[];
  bf16* Et  = (bf16*)smem;                      // MT x KE
  bf16* H1  = (bf16*)(smem + MT * KE * 2);      // MT x HID
  bf16* W1s = H1 + MT * HID;                    // 2 x HID x 32
  bf16* W2s = W1s + 2 * HID * 32;               // 2 x OUT x 32
  int*  nj  = (int*)(W2s + 2 * OUT * 32);       // MT

  const int tid = threadIdx.x;
  const int bb  = blockIdx.x >> 4;
  const int p0  = (blockIdx.x & 15) * 16;

  const int lane = tid & 31;
  const int wave = tid >> 5;
  const int half = lane >> 4;
  const int ml   = lane & 15;
  const int mt   = wave & 3;
  const int wn   = wave >> 2;
  const bool issuer = (wave == 0);

  // Kick off first weight slice while we build the edge tile.
  if (issuer) tdm_load_2d(W1T, (unsigned int)(uintptr_t)W1s, HID, KE);

  if (tid < MT)
    nj[tid] = IDX[((size_t)bb * 256 + p0 + (tid >> 2)) * 4 + (tid & 3)];
  __syncthreads();

  const float* Xb = X + (size_t)bb * 256 * CIN;
  for (int t = tid; t < MT * KE; t += 256) {
    const int e  = t / KE;
    const int kk = t - e * KE;
    const int i  = p0 + (e >> 2);
    float v = 0.f;
    if (kk < CIN) {
      v = Xb[i * CIN + kk];
    } else if (kk < 2 * CIN) {
      const int j  = nj[e];
      const int kc = kk - CIN;
      v = Xb[j * CIN + kc] - Xb[i * CIN + kc];
    }
    Et[t] = (bf16)v;
  }

  const v8f vzero = {0.f, 0.f, 0.f, 0.f, 0.f, 0.f, 0.f, 0.f};

  // ---------------- GEMM1 ----------------
  v8f acc1[NT1W];
#pragma unroll
  for (int j = 0; j < NT1W; ++j) acc1[j] = vzero;

#pragma unroll 1
  for (int ks = 0; ks < NK1; ++ks) {
    if (issuer) __builtin_amdgcn_s_wait_tensorcnt(0);
    __syncthreads();  // publishes staged slice ks (and Et on ks==0)
    if (issuer && (ks + 1) < NK1)
      tdm_load_2d(W1T + (ks + 1) * 32,
                  (unsigned int)(uintptr_t)(W1s + ((ks + 1) & 1) * HID * 32),
                  HID, KE);
    const bf16* Ws = W1s + (ks & 1) * HID * 32;
    const v16bf af = load_a_frag(Et + (mt * 16 + ml) * KE + ks * 32 + half * 16);
#pragma unroll
    for (int j = 0; j < NT1W; ++j) {
      const v16bf bf_ = load_b_frag(Ws + ((wn * NT1W + j) * 16 + ml) * 32 + half * 16);
      acc1[j] = __builtin_amdgcn_wmma_f32_16x16x32_bf16(
          false, af, false, bf_, (short)0, acc1[j], false, false);
    }
  }

  // Pre-stage first W2 slice; overlaps epilogue + barrier.
  if (issuer) tdm_load_2d(W2T, (unsigned int)(uintptr_t)W2s, OUT, HID);

#pragma unroll
  for (int j = 0; j < NT1W; ++j) {
    const int col   = (wn * NT1W + j) * 16 + ml;
    const float bia = B1[col];
#pragma unroll
    for (int v = 0; v < 8; ++v) {
      const float h = leaky_f(acc1[j][v] + bia);
      H1[(mt * 16 + v + half * 8) * HID + col] = (bf16)h;
    }
  }

  // ---------------- GEMM2 + neighbor-sum ----------------
  v8f acc2[NT2W];
#pragma unroll
  for (int j = 0; j < NT2W; ++j) acc2[j] = vzero;

#pragma unroll 1
  for (int ks = 0; ks < NK2; ++ks) {
    if (issuer) __builtin_amdgcn_s_wait_tensorcnt(0);
    __syncthreads();  // publishes W2 slice ks and (ks==0) the H1 writes
    if (issuer && (ks + 1) < NK2)
      tdm_load_2d(W2T + (ks + 1) * 32,
                  (unsigned int)(uintptr_t)(W2s + ((ks + 1) & 1) * OUT * 32),
                  OUT, HID);
    const bf16* Ws = W2s + (ks & 1) * OUT * 32;
    const v16bf af = load_a_frag(H1 + (mt * 16 + ml) * HID + ks * 32 + half * 16);
#pragma unroll
    for (int j = 0; j < NT2W; ++j) {
      const v16bf bf_ = load_b_frag(Ws + ((wn * NT2W + j) * 16 + ml) * 32 + half * 16);
      acc2[j] = __builtin_amdgcn_wmma_f32_16x16x32_bf16(
          false, af, false, bf_, (short)0, acc2[j], false, false);
    }
  }

#pragma unroll
  for (int j = 0; j < NT2W; ++j) {
    const int col   = (wn * NT2W + j) * 16 + ml;
    const float bia = B2[col];
    float s0 = 0.f, s1 = 0.f;
#pragma unroll
    for (int v = 0; v < 4; ++v) s0 += leaky_f(acc2[j][v] + bia);
#pragma unroll
    for (int v = 4; v < 8; ++v) s1 += leaky_f(acc2[j][v] + bia);
    const int pA = p0 + mt * 4 + half * 2;  // rows v+8*half -> point pairs
    Y[((size_t)bb * 256 + pA) * OUT + col]     = s0;
    Y[((size_t)bb * 256 + pA + 1) * OUT + col] = s1;
  }
}

// ---------------------------------------------------------------------------
// Head GEMMs: rows = B*N, A built on the fly from concat [x,a,b,c,d] (K=772
// padded to 800).  leaky after nn1, none after nn2.  Y: (65536 x 192) f32.
// Same TDM-staged weight pipeline as edgeconv.
// ---------------------------------------------------------------------------
__global__ __launch_bounds__(256) void head_gemm_kernel(
    const float* __restrict__ X, const float* __restrict__ A,
    const float* __restrict__ Bm, const float* __restrict__ Cm,
    const float* __restrict__ Dm, const bf16* __restrict__ W1T,
    const float* __restrict__ B1, const bf16* __restrict__ W2T,
    const float* __restrict__ B2, float* __restrict__ Y) {
  constexpr int KE = 800, HID = 256, OUT = 192;
  constexpr int NT1W = (HID / 16) / 2;  // 8
  constexpr int NT2W = (OUT / 16) / 2;  // 6
  constexpr int NK1  = KE / 32;         // 25
  constexpr int NK2  = HID / 32;        // 8

  extern __shared__ char smem[];
  bf16* Et  = (bf16*)smem;                  // 64 x 800
  bf16* H1  = (bf16*)(smem + 64 * KE * 2);  // 64 x 256
  bf16* W1s = H1 + 64 * HID;                // 2 x 256 x 32
  bf16* W2s = W1s + 2 * HID * 32;           // 2 x 192 x 32

  const int tid = threadIdx.x;
  const size_t row0 = (size_t)blockIdx.x * 64;

  const int lane = tid & 31;
  const int wave = tid >> 5;
  const int half = lane >> 4;
  const int ml   = lane & 15;
  const int mt   = wave & 3;
  const int wn   = wave >> 2;
  const bool issuer = (wave == 0);

  if (issuer) tdm_load_2d(W1T, (unsigned int)(uintptr_t)W1s, HID, KE);

  for (int t = tid; t < 64 * KE; t += 256) {
    const int e  = t / KE;
    const int kk = t - e * KE;
    const size_t m = row0 + e;
    float v = 0.f;
    if (kk < 4)        v = X[m * 4 + kk];
    else if (kk < 196) v = A[m * 192 + (kk - 4)];
    else if (kk < 388) v = Bm[m * 192 + (kk - 196)];
    else if (kk < 580) v = Cm[m * 192 + (kk - 388)];
    else if (kk < 772) v = Dm[m * 192 + (kk - 580)];
    Et[t] = (bf16)v;
  }

  const v8f vzero = {0.f, 0.f, 0.f, 0.f, 0.f, 0.f, 0.f, 0.f};

  v8f acc1[NT1W];
#pragma unroll
  for (int j = 0; j < NT1W; ++j) acc1[j] = vzero;

#pragma unroll 1
  for (int ks = 0; ks < NK1; ++ks) {
    if (issuer) __builtin_amdgcn_s_wait_tensorcnt(0);
    __syncthreads();
    if (issuer && (ks + 1) < NK1)
      tdm_load_2d(W1T + (ks + 1) * 32,
                  (unsigned int)(uintptr_t)(W1s + ((ks + 1) & 1) * HID * 32),
                  HID, KE);
    const bf16* Ws = W1s + (ks & 1) * HID * 32;
    const v16bf af = load_a_frag(Et + (mt * 16 + ml) * KE + ks * 32 + half * 16);
#pragma unroll
    for (int j = 0; j < NT1W; ++j) {
      const v16bf bf_ = load_b_frag(Ws + ((wn * NT1W + j) * 16 + ml) * 32 + half * 16);
      acc1[j] = __builtin_amdgcn_wmma_f32_16x16x32_bf16(
          false, af, false, bf_, (short)0, acc1[j], false, false);
    }
  }

  if (issuer) tdm_load_2d(W2T, (unsigned int)(uintptr_t)W2s, OUT, HID);

#pragma unroll
  for (int j = 0; j < NT1W; ++j) {
    const int col   = (wn * NT1W + j) * 16 + ml;
    const float bia = B1[col];
#pragma unroll
    for (int v = 0; v < 8; ++v) {
      const float h = leaky_f(acc1[j][v] + bia);
      H1[(mt * 16 + v + half * 8) * HID + col] = (bf16)h;
    }
  }

  v8f acc2[NT2W];
#pragma unroll
  for (int j = 0; j < NT2W; ++j) acc2[j] = vzero;

#pragma unroll 1
  for (int ks = 0; ks < NK2; ++ks) {
    if (issuer) __builtin_amdgcn_s_wait_tensorcnt(0);
    __syncthreads();
    if (issuer && (ks + 1) < NK2)
      tdm_load_2d(W2T + (ks + 1) * 32,
                  (unsigned int)(uintptr_t)(W2s + ((ks + 1) & 1) * OUT * 32),
                  OUT, HID);
    const bf16* Ws = W2s + (ks & 1) * OUT * 32;
    const v16bf af = load_a_frag(H1 + (mt * 16 + ml) * HID + ks * 32 + half * 16);
#pragma unroll
    for (int j = 0; j < NT2W; ++j) {
      const v16bf bf_ = load_b_frag(Ws + ((wn * NT2W + j) * 16 + ml) * 32 + half * 16);
      acc2[j] = __builtin_amdgcn_wmma_f32_16x16x32_bf16(
          false, af, false, bf_, (short)0, acc2[j], false, false);
    }
  }

#pragma unroll
  for (int j = 0; j < NT2W; ++j) {
    const int col   = (wn * NT2W + j) * 16 + ml;
    const float bia = B2[col];
#pragma unroll
    for (int v = 0; v < 8; ++v) {
      Y[(row0 + mt * 16 + v + half * 8) * OUT + col] = acc2[j][v] + bia;
    }
  }
}

// ---------------------------------------------------------------------------
// Pool: per (batch, channel): max/min/sum/mean over 256 points, then leaky.
// ---------------------------------------------------------------------------
__global__ __launch_bounds__(192) void pool_kernel(const float* __restrict__ H,
                                                   float* __restrict__ P) {
  const int bb = blockIdx.x;
  const int c  = threadIdx.x;  // 0..191
  const float* h = H + (size_t)bb * 256 * 192 + c;
  float mx = -3.402823466e38f, mn = 3.402823466e38f, sm = 0.f;
  for (int p = 0; p < 256; ++p) {
    const float v = h[(size_t)p * 192];
    mx = fmaxf(mx, v); mn = fminf(mn, v); sm += v;
  }
  const float mean = sm * (1.f / 256.f);
  P[(size_t)bb * 768 + 0 * 192 + c] = leaky_f(mx);
  P[(size_t)bb * 768 + 1 * 192 + c] = leaky_f(mn);
  P[(size_t)bb * 768 + 2 * 192 + c] = leaky_f(sm);
  P[(size_t)bb * 768 + 3 * 192 + c] = leaky_f(mean);
}

// ---------------------------------------------------------------------------
// Final tiny MLP: (256 x 768) @ (768 x 96) +b leaky, @ (96 x 1) +b.
// ---------------------------------------------------------------------------
__global__ __launch_bounds__(128) void head_mlp_kernel(
    const float* __restrict__ P, const float* __restrict__ W3,
    const float* __restrict__ B3, const float* __restrict__ W4,
    const float* __restrict__ B4, float* __restrict__ OUT) {
  __shared__ float hbuf[96];
  const int bb = blockIdx.x;
  const int t  = threadIdx.x;
  if (t < 96) {
    float s = B3[t];
    const float* p = P + (size_t)bb * 768;
    for (int k = 0; k < 768; ++k) s += p[k] * W3[(size_t)k * 96 + t];
    hbuf[t] = leaky_f(s);
  }
  __syncthreads();
  if (t == 0) {
    float s = B4[0];
    for (int k = 0; k < 96; ++k) s += hbuf[k] * W4[k];
    OUT[bb] = s;
  }
}

// ---------------------------------------------------------------------------
extern "C" void kernel_launch(void* const* d_in, const int* in_sizes, int n_in,
                              void* d_out, int out_size, void* d_ws,
                              size_t ws_size, hipStream_t stream) {
  const float* x    = (const float*)d_in[0];
  const float* c1w1 = (const float*)d_in[1];
  const float* c1b1 = (const float*)d_in[2];
  const float* c1w2 = (const float*)d_in[3];
  const float* c1b2 = (const float*)d_in[4];
  const float* c2w1 = (const float*)d_in[5];
  const float* c2b1 = (const float*)d_in[6];
  const float* c2w2 = (const float*)d_in[7];
  const float* c2b2 = (const float*)d_in[8];
  const float* c3w1 = (const float*)d_in[9];
  const float* c3b1 = (const float*)d_in[10];
  const float* c3w2 = (const float*)d_in[11];
  const float* c3b2 = (const float*)d_in[12];
  const float* c4w1 = (const float*)d_in[13];
  const float* c4b1 = (const float*)d_in[14];
  const float* c4w2 = (const float*)d_in[15];
  const float* c4b2 = (const float*)d_in[16];
  const float* nn1w = (const float*)d_in[17];
  const float* nn1b = (const float*)d_in[18];
  const float* nn2w = (const float*)d_in[19];
  const float* nn2b = (const float*)d_in[20];
  const float* nn3w = (const float*)d_in[21];
  const float* nn3b = (const float*)d_in[22];
  const float* nn4w = (const float*)d_in[23];
  const float* nn4b = (const float*)d_in[24];

  char* ws = (char*)d_ws;
  size_t off = 0;
  auto take = [&](size_t bytes) -> char* {
    char* p = ws + off;
    off = (off + bytes + 255) & ~(size_t)255;
    return p;
  };

  const size_t ACT = (size_t)256 * 256 * 192 * sizeof(float);
  int*   idxb = (int*)take((size_t)256 * 256 * 4 * sizeof(int));
  float* actA = (float*)take(ACT);
  float* actB = (float*)take(ACT);
  float* actC = (float*)take(ACT);
  float* actD = (float*)take(ACT);
  float* h2   = (float*)take(ACT);
  float* pooled = (float*)take((size_t)256 * 768 * sizeof(float));

  bf16* w1c1T = (bf16*)take((size_t)96 * 32 * 2);
  bf16* w2c1T = (bf16*)take((size_t)192 * 96 * 2);
  bf16* w1T2  = (bf16*)take((size_t)256 * 384 * 2);
  bf16* w2T2  = (bf16*)take((size_t)192 * 256 * 2);
  bf16* w1T3  = (bf16*)take((size_t)256 * 384 * 2);
  bf16* w2T3  = (bf16*)take((size_t)192 * 256 * 2);
  bf16* w1T4  = (bf16*)take((size_t)256 * 384 * 2);
  bf16* w2T4  = (bf16*)take((size_t)192 * 256 * 2);
  bf16* nn1T  = (bf16*)take((size_t)256 * 800 * 2);
  bf16* nn2T  = (bf16*)take((size_t)192 * 256 * 2);
  float* b1p2  = (float*)take(256 * sizeof(float));
  float* b1p3  = (float*)take(256 * sizeof(float));
  float* b1p4  = (float*)take(256 * sizeof(float));
  float* nn1bp = (float*)take(256 * sizeof(float));

  auto packW = [&](const float* W, bf16* WT, int Kin, int Nin, int Kpad,
                   int Npad) {
    const int total = Kpad * Npad;
    pack_wT<<<(total + 255) / 256, 256, 0, stream>>>(W, WT, Kin, Nin, Kpad, Npad);
  };
  packW(c1w1, w1c1T, 8, 96, 32, 96);
  packW(c1w2, w2c1T, 96, 192, 96, 192);
  packW(c2w1, w1T2, 384, 252, 384, 256);
  packW(c2w2, w2T2, 252, 192, 256, 192);
  packW(c3w1, w1T3, 384, 252, 384, 256);
  packW(c3w2, w2T3, 252, 192, 256, 192);
  packW(c4w1, w1T4, 384, 252, 384, 256);
  packW(c4w2, w2T4, 252, 192, 256, 192);
  packW(nn1w, nn1T, 772, 252, 800, 256);
  packW(nn2w, nn2T, 252, 192, 256, 192);
  pack_bias<<<1, 256, 0, stream>>>(c2b1, b1p2, 252, 256);
  pack_bias<<<1, 256, 0, stream>>>(c3b1, b1p3, 252, 256);
  pack_bias<<<1, 256, 0, stream>>>(c4b1, b1p4, 252, 256);
  pack_bias<<<1, 256, 0, stream>>>(nn1b, nn1bp, 252, 256);

  // dynamic LDS: Et + H1 + 2x(HIDx32) + 2x(OUTx32) staged weights + nj
  const size_t lds1 = (size_t)64 * 32 * 2 + (size_t)64 * 96 * 2 +
                      2 * (size_t)96 * 64 + 2 * (size_t)192 * 64 + 256;
  const size_t lds2 = (size_t)64 * 384 * 2 + (size_t)64 * 256 * 2 +
                      2 * (size_t)256 * 64 + 2 * (size_t)192 * 64 + 256;
  const size_t ldsH = (size_t)64 * 800 * 2 + (size_t)64 * 256 * 2 +
                      2 * (size_t)256 * 64 + 2 * (size_t)192 * 64;

  knn_kernel<<<256, 256, 0, stream>>>(x, 4, idxb);
  edgeconv_kernel<4, 32, 96, 192>
      <<<4096, 256, lds1, stream>>>(x, idxb, w1c1T, c1b1, w2c1T, c1b2, actA);

  knn_kernel<<<256, 256, 0, stream>>>(actA, 192, idxb);
  edgeconv_kernel<192, 384, 256, 192>
      <<<4096, 256, lds2, stream>>>(actA, idxb, w1T2, b1p2, w2T2, c2b2, actB);

  knn_kernel<<<256, 256, 0, stream>>>(actB, 192, idxb);
  edgeconv_kernel<192, 384, 256, 192>
      <<<4096, 256, lds2, stream>>>(actB, idxb, w1T3, b1p3, w2T3, c3b2, actC);

  knn_kernel<<<256, 256, 0, stream>>>(actC, 192, idxb);
  edgeconv_kernel<192, 384, 256, 192>
      <<<4096, 256, lds2, stream>>>(actC, idxb, w1T4, b1p4, w2T4, c4b2, actD);

  head_gemm_kernel<<<1024, 256, ldsH, stream>>>(x, actA, actB, actC, actD, nn1T,
                                                nn1bp, nn2T, nn2b, h2);
  pool_kernel<<<256, 192, 0, stream>>>(h2, pooled);
  head_mlp_kernel<<<256, 128, 0, stream>>>(pooled, nn3w, nn3b, nn4w, nn4b,
                                           (float*)d_out);
  (void)in_sizes; (void)n_in; (void)out_size; (void)ws_size;
}
